// trajectory2seq_13469017440301
// MI455X (gfx1250) — compile-verified, matrix operand order checked by reading, not run on hardware
//
#include <hip/hip_runtime.h>
#include <hip/hip_bf16.h>

typedef _Float16 v16h __attribute__((ext_vector_type(16)));
typedef float    v8f  __attribute__((ext_vector_type(8)));

static __device__ __forceinline__ v8f wmma16(v16h a, v16h b, v8f c) {
    // D = A(16x32 f16) * B(32x16 f16) + C(16x16 f32)
    return __builtin_amdgcn_wmma_f32_16x16x32_f16(false, a, false, b, (short)0, c, false, false);
}
static __device__ __forceinline__ float sigf(float v) {
    return 1.0f / (1.0f + __expf(-v));
}

// ---------------------------------------------------------------- constants
#define BB   128
#define TT   512
#define HH   256
#define H2   512
#define LL   64
#define DICT 30
#define ROWP 272   // padded LDS row (halves); 272*2 bytes = 544, 32B aligned

// ---------------------------------------------------------------- f32 -> f16
__global__ void cvt_f16(const float* __restrict__ src, _Float16* __restrict__ dst, int n) {
    int i = blockIdx.x * blockDim.x + threadIdx.x;
    if (i < n) dst[i] = (_Float16)src[i];
}

// ---------------------------------------------------------------- encoder GRU
// 2 persistent workgroups (blockIdx.x = direction), 512 threads = 16 waves,
// i.e. exactly 4 waves/SIMD on one WGP -> __launch_bounds__(512,1) so the
// compiler budgets VGPRs for that occupancy instead of spilling accumulators.
// wave w owns gate columns c in [16w,16w+16): r=c, z=256+c, n=512+c of gh, and
// therefore owns h_new[:, c].  h ping-pongs in LDS (f16); Whh streams from L2.
// Loop nest: m-half (4 m-tiles of accumulators, 96 VGPRs) -> kt (B tiles live
// <= 2 iters) -> 4 m-tiles.
__global__ void __launch_bounds__(512, 1)
enc_gru(const float* __restrict__ x,
        const _Float16* __restrict__ WhhF, const _Float16* __restrict__ WhhB,
        const float* __restrict__ WihF, const float* __restrict__ WihB,
        const float* __restrict__ bihF, const float* __restrict__ bhhF,
        const float* __restrict__ bihB, const float* __restrict__ bhhB,
        _Float16* __restrict__ enc_out,   // [B][T][2H] f16
        _Float16* __restrict__ hdec0)     // [B][2H]  decoder h0 (f16)
{
    const int dir = blockIdx.x;
    const _Float16* Whh = dir ? WhhB : WhhF;
    const float* Wih = dir ? WihB : WihF;
    const float* bih = dir ? bihB : bihF;
    const float* bhh = dir ? bhhB : bhhF;

    extern __shared__ _Float16 hlds[];   // [2][128][ROWP]

    const int tid  = threadIdx.x;
    const int w    = tid >> 5;           // wave 0..15
    const int lane = tid & 31;
    const int l15  = lane & 15;
    const int hi   = lane >> 4;
    const int c    = 16 * w + l15;       // owned h column, 0..255

    // B-tile row bases for this lane (r/z/n gate rows of Whh)
    const _Float16* bRrow = Whh + (size_t)(c)       * HH + hi * 16;
    const _Float16* bZrow = Whh + (size_t)(256 + c) * HH + hi * 16;
    const _Float16* bNrow = Whh + (size_t)(512 + c) * HH + hi * 16;

    // zero h buffer 0
    for (int i = tid; i < BB * ROWP; i += 512) hlds[i] = (_Float16)0.0f;
    __syncthreads();

    // per-lane constant gate parameters (column-resident across all T).
    // r/z gates: bih+bhh always appear summed -> fold into one constant.
    const float wr0 = Wih[(c)*2 + 0],      wr1 = Wih[(c)*2 + 1];
    const float wz0 = Wih[(256 + c)*2],    wz1 = Wih[(256 + c)*2 + 1];
    const float wn0 = Wih[(512 + c)*2],    wn1 = Wih[(512 + c)*2 + 1];
    const float bR2 = bih[c] + bhh[c];
    const float bZ2 = bih[256 + c] + bhh[256 + c];
    const float bnI = bih[512 + c], bnH = bhh[512 + c];

    for (int t = 0; t < TT; ++t) {
        const int p  = t & 1;
        const _Float16* hprev = hlds + p * (BB * ROWP);
        _Float16*       hnext = hlds + (p ^ 1) * (BB * ROWP);
        const int tx = dir ? (TT - 1 - t) : t;

        // prefetch next step's input rows (L2 hint)
        if (t + 1 < TT) {
            int txn = dir ? (TT - 2 - t) : (t + 1);
            __builtin_prefetch(&x[((size_t)(tid & 127) * TT + txn) * 2], 0, 1);
        }

        for (int half = 0; half < 2; ++half) {
            v8f aR[4] = {}, aZ[4] = {}, aN[4] = {};
#pragma unroll 2
            for (int kt = 0; kt < 8; ++kt) {
                const int koff = kt * 32;
                const v16h bR = *(const v16h*)(bRrow + koff);
                const v16h bZ = *(const v16h*)(bZrow + koff);
                const v16h bN = *(const v16h*)(bNrow + koff);
#pragma unroll
                for (int q = 0; q < 4; ++q) {
                    const int mt = half * 4 + q;
                    const v16h A = *(const v16h*)(hprev + (size_t)(16 * mt + l15) * ROWP
                                                  + koff + hi * 16);
                    aR[q] = wmma16(A, bR, aR[q]);
                    aZ[q] = wmma16(A, bZ, aZ[q]);
                    aN[q] = wmma16(A, bN, aN[q]);
                }
            }
#pragma unroll
            for (int q = 0; q < 4; ++q) {
                const int mt = half * 4 + q;
#pragma unroll
                for (int j = 0; j < 8; ++j) {
                    const int m = 16 * mt + 8 * hi + j;      // batch row
                    const float2 xv = ((const float2*)x)[(size_t)m * TT + tx];
                    const float r  = sigf(xv.x * wr0 + xv.y * wr1 + bR2 + aR[q][j]);
                    const float z  = sigf(xv.x * wz0 + xv.y * wz1 + bZ2 + aZ[q][j]);
                    const float nn = tanhf(xv.x * wn0 + xv.y * wn1 + bnI + r * (aN[q][j] + bnH));
                    const float hold = (float)hprev[m * ROWP + c];
                    const float hnew = (1.0f - z) * nn + z * hold;
                    hnext[m * ROWP + c] = (_Float16)hnew;
                    enc_out[((size_t)m * TT + tx) * H2 + dir * HH + c] = (_Float16)hnew;
                    if (t == TT - 1) hdec0[m * H2 + dir * HH + c] = (_Float16)hnew;
                }
            }
        }
        __syncthreads();
    }
}

// ---------------------------------------------------------------- decoder GRU (one step)
// 32 blocks x 8 waves. block nb owns gate columns c in [16nb,16nb+16) of the 512-wide
// hidden; wave w owns batch tile m in [16w,16w+16).  A-tiles gather token embeddings.
__global__ void dec_gru(int s, const int* __restrict__ tgt,
                        const _Float16* __restrict__ emb16,
                        const _Float16* __restrict__ Wih16,   // [1536][256]
                        const _Float16* __restrict__ Whh16,   // [1536][512]
                        const float* __restrict__ bih, const float* __restrict__ bhh,
                        const _Float16* __restrict__ hprev,   // [128][512]
                        _Float16* __restrict__ hnext, float* __restrict__ h32)
{
    const int nb   = blockIdx.x;
    const int tid  = threadIdx.x;
    const int w    = tid >> 5;
    const int l15  = tid & 15;
    const int hi   = (tid & 31) >> 4;
    const int c    = 16 * nb + l15;      // 0..511
    const int rowm = 16 * w + l15;       // A-tile row (batch)
    const int tok  = (s == 0) ? 1 : tgt[rowm * LL + (s - 1)];

    v8f aR = {}, aZ = {}, aIN = {}, aHN = {};
    // part 1: emb[tok] @ dec_Wih.T  (K = 256)
#pragma unroll 2
    for (int kt = 0; kt < 8; ++kt) {
        const int koff = kt * 32 + hi * 16;
        v16h A  = *(const v16h*)(emb16 + (size_t)tok * HH + koff);
        v16h bR = *(const v16h*)(Wih16 + (size_t)(c)        * HH + koff);
        v16h bZ = *(const v16h*)(Wih16 + (size_t)(512 + c)  * HH + koff);
        v16h bN = *(const v16h*)(Wih16 + (size_t)(1024 + c) * HH + koff);
        aR = wmma16(A, bR, aR); aZ = wmma16(A, bZ, aZ); aIN = wmma16(A, bN, aIN);
    }
    // part 2: h @ dec_Whh.T  (K = 512)
#pragma unroll 2
    for (int kt = 0; kt < 16; ++kt) {
        const int koff = kt * 32 + hi * 16;
        v16h A  = *(const v16h*)(hprev + (size_t)rowm * H2 + koff);
        v16h bR = *(const v16h*)(Whh16 + (size_t)(c)        * H2 + koff);
        v16h bZ = *(const v16h*)(Whh16 + (size_t)(512 + c)  * H2 + koff);
        v16h bN = *(const v16h*)(Whh16 + (size_t)(1024 + c) * H2 + koff);
        aR = wmma16(A, bR, aR); aZ = wmma16(A, bZ, aZ); aHN = wmma16(A, bN, aHN);
    }
    const float bR2 = bih[c] + bhh[c];
    const float bZ2 = bih[512 + c] + bhh[512 + c];
    const float bnI = bih[1024 + c], bnH = bhh[1024 + c];
#pragma unroll
    for (int j = 0; j < 8; ++j) {
        const int m = 16 * w + 8 * hi + j;
        const float r  = sigf(aR[j] + bR2);
        const float z  = sigf(aZ[j] + bZ2);
        const float nn = tanhf(aIN[j] + bnI + r * (aHN[j] + bnH));
        const float hold = (float)hprev[m * H2 + c];
        const float hnew = (1.0f - z) * nn + z * hold;
        hnext[m * H2 + c] = (_Float16)hnew;
        h32[m * H2 + c]   = hnew;
    }
}

// ---------------------------------------------------------------- attention (one step)
// block per batch element; q = h@Wq.T+bq, softmax over T, ctx accumulation.
__global__ void dec_attn(int s,
                         const _Float16* __restrict__ hcur,   // [128][512]
                         const float* __restrict__ Wq, const float* __restrict__ bq,
                         const _Float16* __restrict__ enc_out, // [B][T][512]
                         _Float16* __restrict__ ctx16,        // [128][512]
                         float* __restrict__ attn_out)        // [B][T][L]
{
    const int b = blockIdx.x, tid = threadIdx.x;
    __shared__ float hsh[H2];
    __shared__ float qsh[H2];
    __shared__ float ssh[TT];
    __shared__ float red[256];

    for (int k = tid; k < H2; k += 256) hsh[k] = (float)hcur[b * H2 + k];
    __syncthreads();

    for (int j = tid; j < H2; j += 256) {
        float a = bq[j];
        const float* wr = Wq + (size_t)j * H2;
        for (int k = 0; k < H2; ++k) a += hsh[k] * wr[k];
        qsh[j] = a;
    }
    __syncthreads();

    for (int t = tid; t < TT; t += 256) {
        const _Float16* e = enc_out + ((size_t)b * TT + t) * H2;
        float a = 0.0f;
        for (int k = 0; k < H2; ++k) a += qsh[k] * (float)e[k];
        ssh[t] = a;
    }
    __syncthreads();

    // softmax: max reduce
    red[tid] = fmaxf(ssh[tid], ssh[tid + 256]);
    __syncthreads();
    for (int st = 128; st > 0; st >>= 1) {
        if (tid < st) red[tid] = fmaxf(red[tid], red[tid + st]);
        __syncthreads();
    }
    const float mx = red[0];
    __syncthreads();
    // exp + sum reduce
    const float e0 = __expf(ssh[tid] - mx), e1 = __expf(ssh[tid + 256] - mx);
    ssh[tid] = e0; ssh[tid + 256] = e1;
    red[tid] = e0 + e1;
    __syncthreads();
    for (int st = 128; st > 0; st >>= 1) {
        if (tid < st) red[tid] += red[tid + st];
        __syncthreads();
    }
    const float inv = 1.0f / red[0];
    __syncthreads();

    for (int t = tid; t < TT; t += 256) {
        const float wv = ssh[t] * inv;
        ssh[t] = wv;
        attn_out[((size_t)b * TT + t) * LL + s] = wv;
    }
    __syncthreads();

    // ctx: coalesced accumulation over T
    float c0 = 0.0f, c1 = 0.0f;
    for (int t = 0; t < TT; ++t) {
        const _Float16* e = enc_out + ((size_t)b * TT + t) * H2;
        const float wv = ssh[t];
        c0 += wv * (float)e[tid];
        c1 += wv * (float)e[tid + 256];
    }
    ctx16[b * H2 + tid]       = (_Float16)c0;
    ctx16[b * H2 + tid + 256] = (_Float16)c1;
}

// ---------------------------------------------------------------- att_combine (one step)
// o = [h | ctx] @ Wc.T + bc  -> [128][256] f16.  16 blocks x 8 waves, WMMA K=1024.
__global__ void dec_comb(const _Float16* __restrict__ hcur, const _Float16* __restrict__ ctx16,
                         const _Float16* __restrict__ Wc16,  // [256][1024]
                         const float* __restrict__ bc, _Float16* __restrict__ o16)
{
    const int nb  = blockIdx.x;
    const int tid = threadIdx.x;
    const int w   = tid >> 5;
    const int l15 = tid & 15;
    const int hi  = (tid & 31) >> 4;
    const int c   = 16 * nb + l15;      // 0..255
    v8f acc = {};
#pragma unroll 2
    for (int kt = 0; kt < 32; ++kt) {
        const int koff = kt * 32 + hi * 16;
        v16h A = (koff < 512)
            ? *(const v16h*)(hcur  + (size_t)(16 * w + l15) * H2 + koff)
            : *(const v16h*)(ctx16 + (size_t)(16 * w + l15) * H2 + (koff - 512));
        v16h Bv = *(const v16h*)(Wc16 + (size_t)c * 1024 + koff);
        acc = wmma16(A, Bv, acc);
    }
    const float bcv = bc[c];
#pragma unroll
    for (int j = 0; j < 8; ++j) {
        const int m = 16 * w + 8 * hi + j;
        o16[m * HH + c] = (_Float16)(acc[j] + bcv);
    }
}

// ---------------------------------------------------------------- fc (one step)
__global__ void dec_fc(int s, const _Float16* __restrict__ o16,
                       const float* __restrict__ Wf, const float* __restrict__ bf,
                       float* __restrict__ vec_out)
{
    const int b = blockIdx.x, d = threadIdx.x;
    if (d < DICT) {
        float a = bf[d];
        for (int k = 0; k < HH; ++k) a += (float)o16[b * HH + k] * Wf[d * HH + k];
        vec_out[((size_t)b * LL + s) * DICT + d] = a;
    }
}

// ---------------------------------------------------------------- launch
extern "C" void kernel_launch(void* const* d_in, const int* in_sizes, int n_in,
                              void* d_out, int out_size, void* d_ws, size_t ws_size,
                              hipStream_t stream) {
    const float* x   = (const float*)d_in[0];
    const int*   tgt = (const int*)d_in[1];

    float* out      = (float*)d_out;
    float* vec_out  = out;                       // 128*64*30 = 245760
    float* hT_out   = out + 245760;              // 128*512   = 65536
    float* attn_out = out + 245760 + 65536;      // 128*512*64

    char* p = (char*)d_ws;
    auto take = [&](size_t bytes) -> char* {
        char* q = p; p += (bytes + 255) & ~(size_t)255; return q;
    };
    _Float16* WhhF  = (_Float16*)take((size_t)196608 * 2);
    _Float16* WhhB  = (_Float16*)take((size_t)196608 * 2);
    _Float16* dWih  = (_Float16*)take((size_t)393216 * 2);
    _Float16* dWhh  = (_Float16*)take((size_t)786432 * 2);
    _Float16* Wc16  = (_Float16*)take((size_t)262144 * 2);
    _Float16* emb16 = (_Float16*)take((size_t)7680 * 2);
    _Float16* encO  = (_Float16*)take((size_t)33554432 * 2);  // [128][512][512] f16
    _Float16* hb0   = (_Float16*)take((size_t)65536 * 2);
    _Float16* hb1   = (_Float16*)take((size_t)65536 * 2);
    _Float16* ctx16 = (_Float16*)take((size_t)65536 * 2);
    _Float16* o16   = (_Float16*)take((size_t)32768 * 2);
    float*    h32   = (float*)take((size_t)65536 * 4);

    auto cvt = [&](const void* s, _Float16* d, int n) {
        cvt_f16<<<(n + 255) / 256, 256, 0, stream>>>((const float*)s, d, n);
    };
    cvt(d_in[3],  WhhF, 196608);
    cvt(d_in[7],  WhhB, 196608);
    cvt(d_in[11], dWih, 393216);
    cvt(d_in[12], dWhh, 786432);
    cvt(d_in[17], Wc16, 262144);
    cvt(d_in[10], emb16, 7680);

    const size_t ldsBytes = (size_t)2 * BB * ROWP * sizeof(_Float16);  // 139264
    hipFuncSetAttribute(reinterpret_cast<const void*>(enc_gru),
                        hipFuncAttributeMaxDynamicSharedMemorySize, (int)ldsBytes);
    enc_gru<<<2, 512, ldsBytes, stream>>>(
        x, WhhF, WhhB,
        (const float*)d_in[2], (const float*)d_in[6],
        (const float*)d_in[4], (const float*)d_in[5],
        (const float*)d_in[8], (const float*)d_in[9],
        encO, hb0);

    for (int s = 0; s < LL; ++s) {
        const _Float16* hp = (s & 1) ? hb1 : hb0;
        _Float16*       hn = (s & 1) ? hb0 : hb1;
        dec_gru<<<32, 256, 0, stream>>>(s, tgt, emb16, dWih, dWhh,
                                        (const float*)d_in[13], (const float*)d_in[14],
                                        hp, hn, h32);
        dec_attn<<<128, 256, 0, stream>>>(s, hn, (const float*)d_in[15],
                                          (const float*)d_in[16], encO, ctx16, attn_out);
        dec_comb<<<16, 256, 0, stream>>>(hn, ctx16, Wc16, (const float*)d_in[18], o16);
        dec_fc<<<128, 32, 0, stream>>>(s, o16, (const float*)d_in[19],
                                       (const float*)d_in[20], vec_out);
    }
    hipMemcpyAsync(hT_out, h32, (size_t)65536 * 4, hipMemcpyDeviceToDevice, stream);
}
